// SparseToDenseAttn_2413771621109
// MI455X (gfx1250) — matrix operand order repaired and checked by reading, not compiled, not run on hardware
//
#include <hip/hip_runtime.h>

// ---------------------------------------------------------------------------
// SparseToDenseAttn fused kernel for gfx1250 (MI455X, wave32, WMMA)
//   out = softmax_ch( f_g( f_q(Wq X) - f_k(Wk sp) ) ) * f_v(Wv X)
//   f_* = BN(inference) + LeakyReLU(0.2), all 1x1 convs == 256x256 GEMMs.
// GEMM path: v_wmma_f32_16x16x32_bf16, bf16 operands / f32 accumulate.
// ---------------------------------------------------------------------------

typedef __attribute__((ext_vector_type(16))) __bf16 v16bf;
typedef __attribute__((ext_vector_type(8)))  float  v8f;

union Frag { v16bf v; unsigned int u[8]; };

#define EMB        256
#define PTS_HALF   128
#define LDS_STRIDE 264   // 256 + 8 pad -> 528B row = 132 dwords = 4-bank lane step
#define SMEM_BYTES (2 * PTS_HALF * LDS_STRIDE * 2 + 512 * 4 + 128 * 4)  // 137728

__device__ __forceinline__ unsigned short f2bf(float f) {
  unsigned int u = __float_as_uint(f);
  unsigned int r = u + 0x7FFFu + ((u >> 16) & 1u);   // round-to-nearest-even
  return (unsigned short)(r >> 16);
}

__device__ __forceinline__ float lrelu(float y) { return y >= 0.f ? y : 0.2f * y; }

// A fragment (16x32 bf16, row = out-channel, K = in-channel), W row-major.
// Lanes 0-15: K=0..15 (2/VGPR); lanes 16-31: K=16..31.
__device__ __forceinline__ void load_a_frag(const unsigned short* __restrict__ W,
                                            int row0, int k0, int lane, Frag& f) {
  const unsigned short* base =
      W + (size_t)(row0 + (lane & 15)) * EMB + k0 + ((lane >> 4) << 4);
#pragma unroll
  for (int j = 0; j < 4; ++j) f.u[j]     = *(const unsigned int*)(base + 2 * j);
#pragma unroll
  for (int j = 0; j < 4; ++j) f.u[4 + j] = *(const unsigned int*)(base + 8 + 2 * j);
}

// B fragment (32x16 bf16, K x points) read from transposed LDS tile [p][c].
// Lanes 0-15: K=0..15 (2/VGPR); lanes 16-31: K=16..31.
__device__ __forceinline__ void load_b_frag(const unsigned short* B,
                                            int p0, int k0, int lane, Frag& f) {
  const unsigned short* base =
      B + (p0 + (lane & 15)) * LDS_STRIDE + k0 + ((lane >> 4) << 4);
#pragma unroll
  for (int j = 0; j < 8; ++j) f.u[j] = *(const unsigned int*)(base + 2 * j);
}

// 64(out) x 32(pts) x 256(K) per wave: acc[rt 0..3][ct 0..1], 8 K-steps.
__device__ __forceinline__ void gemm_tile(const unsigned short* __restrict__ W,
                                          const unsigned short* B,
                                          int rowBase, int colBase, int lane,
                                          v8f acc[4][2]) {
#pragma unroll
  for (int rt = 0; rt < 4; ++rt)
#pragma unroll
    for (int ct = 0; ct < 2; ++ct)
#pragma unroll
      for (int i = 0; i < 8; ++i) acc[rt][ct][i] = 0.f;

  for (int k0 = 0; k0 < EMB; k0 += 32) {
    Frag b0, b1;
    load_b_frag(B, colBase,      k0, lane, b0);
    load_b_frag(B, colBase + 16, k0, lane, b1);
#pragma unroll
    for (int rt = 0; rt < 4; ++rt) {
      Frag a;
      load_a_frag(W, rowBase + rt * 16, k0, lane, a);
      acc[rt][0] = __builtin_amdgcn_wmma_f32_16x16x32_bf16(
          false, a.v, false, b0.v, (short)0, acc[rt][0], false, false);
      acc[rt][1] = __builtin_amdgcn_wmma_f32_16x16x32_bf16(
          false, a.v, false, b1.v, (short)0, acc[rt][1], false, false);
    }
  }
}

// --------------------------- prologue kernels ------------------------------

// Convert Wq/Wv/Wg (256x256 f32 row-major) to bf16 in workspace.
__global__ __launch_bounds__(256) void conv_weights_kernel(
    const float* __restrict__ wq, const float* __restrict__ wv,
    const float* __restrict__ wg, unsigned short* __restrict__ dst) {
  int i = blockIdx.x * 256 + threadIdx.x;          // 0 .. 196607
  if (i < 65536)        dst[i] = f2bf(wq[i]);
  else if (i < 131072)  dst[i] = f2bf(wv[i - 65536]);
  else                  dst[i] = f2bf(wg[i - 131072]);
}

// k-vector per (batch, stroke): k[o] = lrelu(BN(Wk . sparse[b,:,s]))
__global__ __launch_bounds__(256) void kvec_kernel(
    const float* __restrict__ sparse, const float* __restrict__ wk,
    const float* __restrict__ gk, const float* __restrict__ bk,
    const float* __restrict__ mk, const float* __restrict__ vk,
    float* __restrict__ kval) {
  __shared__ float sp[EMB];
  const int bs = blockIdx.x;                       // b*32 + s
  const int b = bs >> 5, s = bs & 31;
  const int o = threadIdx.x;
  sp[o] = sparse[((size_t)b * EMB + o) * 32 + s];
  __syncthreads();
  const float* wrow = wk + (size_t)o * EMB;
  float dot = 0.f;
#pragma unroll 4
  for (int c = 0; c < EMB; ++c) dot += wrow[c] * sp[c];
  float a = gk[o] * rsqrtf(vk[o] + 1e-5f);
  float y = a * dot + (bk[o] - mk[o] * a);
  kval[(size_t)bs * EMB + o] = lrelu(y);
}

// ------------------------------ main kernel --------------------------------

__global__ __launch_bounds__(512, 1) void attn_main_kernel(
    const float* __restrict__ dense,
    const unsigned short* __restrict__ wqb, const unsigned short* __restrict__ wvb,
    const unsigned short* __restrict__ wgb,
    const float* __restrict__ gq, const float* __restrict__ bq,
    const float* __restrict__ mq, const float* __restrict__ vq,
    const float* __restrict__ gv, const float* __restrict__ bv,
    const float* __restrict__ mv, const float* __restrict__ vv,
    const float* __restrict__ gg, const float* __restrict__ bg,
    const float* __restrict__ mg, const float* __restrict__ vg,
    const float* __restrict__ kval, float* __restrict__ out) {
  extern __shared__ char smem[];
  unsigned short* XT = (unsigned short*)smem;                              // [128][264] bf16
  unsigned short* TT = (unsigned short*)(smem + PTS_HALF * LDS_STRIDE * 2);// [128][264] bf16
  float* red  = (float*)(smem + 2 * PTS_HALF * LDS_STRIDE * 2);            // [4][128]
  float* colv = red + 512;                                                 // [128]

  const int tid  = threadIdx.x;
  const int lane = tid & 31;
  const int w    = tid >> 5;
  const int wm   = w & 3, wn = w >> 2;
  const int rowBase = wm * 64, colBase = wn * 32;
  const int nloc = lane & 15, hl = lane >> 4;

  const int tileId = blockIdx.x;       // 0..1023
  const int h  = tileId & 1;
  const int bs = tileId >> 1;          // b*32 + s
  const int b  = bs >> 5, s = bs & 31;

  const float* xg = dense + ((size_t)b * EMB) * 8192 + s * 256 + h * PTS_HALF;

  // ---- stage X^T (bf16) into LDS: coalesced float4 reads over points ----
#pragma unroll
  for (int it = 0; it < 16; ++it) {
    int idx = it * 512 + tid;          // 0..8191
    int c   = idx >> 5;                // channel
    int p4  = (idx & 31) << 2;         // point (step 4)
    const float4 x4 = *(const float4*)(xg + (size_t)c * 8192 + p4);
    XT[(p4 + 0) * LDS_STRIDE + c] = f2bf(x4.x);
    XT[(p4 + 1) * LDS_STRIDE + c] = f2bf(x4.y);
    XT[(p4 + 2) * LDS_STRIDE + c] = f2bf(x4.z);
    XT[(p4 + 3) * LDS_STRIDE + c] = f2bf(x4.w);
  }
  __syncthreads();

  v8f acc[4][2];

  // ---- phase 1: Q = lrelu(BN(Wq X)); T = Q - k -> TT (bf16) ----
  gemm_tile(wqb, XT, rowBase, colBase, lane, acc);
  const float* kv = kval + (size_t)bs * EMB;
#pragma unroll
  for (int rt = 0; rt < 4; ++rt) {
    const int ob = rowBase + rt * 16 + hl * 8;     // 8 consecutive channels
    float av[8], cv[8], kk[8];
#pragma unroll
    for (int r = 0; r < 8; ++r) {
      float a = gq[ob + r] * rsqrtf(vq[ob + r] + 1e-5f);
      av[r] = a; cv[r] = bq[ob + r] - mq[ob + r] * a; kk[r] = kv[ob + r];
    }
#pragma unroll
    for (int ct = 0; ct < 2; ++ct) {
      int p = colBase + ct * 16 + nloc;
      unsigned int* dst = (unsigned int*)(TT + p * LDS_STRIDE + ob);
#pragma unroll
      for (int r = 0; r < 8; r += 2) {
        float t0 = lrelu(acc[rt][ct][r] * av[r] + cv[r]) - kk[r];
        float t1 = lrelu(acc[rt][ct][r + 1] * av[r + 1] + cv[r + 1]) - kk[r + 1];
        dst[r >> 1] = (unsigned)f2bf(t0) | ((unsigned)f2bf(t1) << 16);
      }
    }
  }
  __syncthreads();

  // ---- phase 2: G = lrelu(BN(Wg T)); softmax over the 256 channels ----
  gemm_tile(wgb, TT, rowBase, colBase, lane, acc);
#pragma unroll
  for (int rt = 0; rt < 4; ++rt) {
    const int ob = rowBase + rt * 16 + hl * 8;
#pragma unroll
    for (int r = 0; r < 8; ++r) {
      float a = gg[ob + r] * rsqrtf(vg[ob + r] + 1e-5f);
      float c = bg[ob + r] - mg[ob + r] * a;
      acc[rt][0][r] = lrelu(acc[rt][0][r] * a + c);
      acc[rt][1][r] = lrelu(acc[rt][1][r] * a + c);
    }
  }
  // per-column max (channels split over lane-halves + 4 row-waves)
#pragma unroll
  for (int ct = 0; ct < 2; ++ct) {
    float m = -3.0e38f;
#pragma unroll
    for (int rt = 0; rt < 4; ++rt)
#pragma unroll
      for (int r = 0; r < 8; ++r) m = fmaxf(m, acc[rt][ct][r]);
    m = fmaxf(m, __shfl_xor(m, 16, 32));
    if (hl == 0) red[wm * 128 + colBase + ct * 16 + nloc] = m;
  }
  __syncthreads();
  if (tid < 128)
    colv[tid] = fmaxf(fmaxf(red[tid], red[128 + tid]),
                      fmaxf(red[256 + tid], red[384 + tid]));
  __syncthreads();
#pragma unroll
  for (int ct = 0; ct < 2; ++ct) {
    int p = colBase + ct * 16 + nloc;
    float cm = colv[p], ssum = 0.f;
#pragma unroll
    for (int rt = 0; rt < 4; ++rt)
#pragma unroll
      for (int r = 0; r < 8; ++r) {
        float e = __expf(acc[rt][ct][r] - cm);
        acc[rt][ct][r] = e; ssum += e;
      }
    ssum += __shfl_xor(ssum, 16, 32);
    if (hl == 0) red[wm * 128 + p] = ssum;
  }
  __syncthreads();
  if (tid < 128)
    colv[tid] = red[tid] + red[128 + tid] + red[256 + tid] + red[384 + tid];
  __syncthreads();
  // coef = e / colsum -> overwrite TT (bf16); T fully consumed already
#pragma unroll
  for (int ct = 0; ct < 2; ++ct) {
    int p = colBase + ct * 16 + nloc;
    float inv = 1.0f / colv[p];
#pragma unroll
    for (int rt = 0; rt < 4; ++rt) {
      const int ob = rowBase + rt * 16 + hl * 8;
      unsigned int* dst = (unsigned int*)(TT + p * LDS_STRIDE + ob);
#pragma unroll
      for (int r = 0; r < 8; r += 2)
        dst[r >> 1] = (unsigned)f2bf(acc[rt][ct][r] * inv) |
                      ((unsigned)f2bf(acc[rt][ct][r + 1] * inv) << 16);
    }
  }
  __syncthreads();

  // ---- phase 3: V = lrelu(BN(Wv X)); out = coef * V ----
  gemm_tile(wvb, XT, rowBase, colBase, lane, acc);
#pragma unroll
  for (int rt = 0; rt < 4; ++rt) {
    const int ob = rowBase + rt * 16 + hl * 8;
    float av[8], cv[8];
#pragma unroll
    for (int r = 0; r < 8; ++r) {
      float a = gv[ob + r] * rsqrtf(vv[ob + r] + 1e-5f);
      av[r] = a; cv[r] = bv[ob + r] - mv[ob + r] * a;
    }
#pragma unroll
    for (int ct = 0; ct < 2; ++ct) {
      int p = colBase + ct * 16 + nloc;
      const unsigned int* csrc = (const unsigned int*)(TT + p * LDS_STRIDE + ob);
      float* og = out + ((size_t)(b * EMB + ob)) * 8192 + s * 256 + h * PTS_HALF + p;
#pragma unroll
      for (int r = 0; r < 8; r += 2) {
        unsigned int cu = csrc[r >> 1];
        float c0 = __uint_as_float(cu << 16);            // low bf16
        float c1 = __uint_as_float(cu & 0xFFFF0000u);    // high bf16
        float y0 = lrelu(acc[rt][ct][r] * av[r] + cv[r]) * c0;
        float y1 = lrelu(acc[rt][ct][r + 1] * av[r + 1] + cv[r + 1]) * c1;
        og[(size_t)r * 8192]       = y0;                 // 16 lanes coalesced per row
        og[(size_t)(r + 1) * 8192] = y1;
      }
    }
  }
}

// ------------------------------- launcher ----------------------------------

extern "C" void kernel_launch(void* const* d_in, const int* in_sizes, int n_in,
                              void* d_out, int out_size, void* d_ws, size_t ws_size,
                              hipStream_t stream) {
  const float* sparse = (const float*)d_in[0];
  const float* dense  = (const float*)d_in[1];
  const float* wq = (const float*)d_in[2];
  const float* gq = (const float*)d_in[3];
  const float* bq = (const float*)d_in[4];
  const float* mq = (const float*)d_in[5];
  const float* vq = (const float*)d_in[6];
  const float* wk = (const float*)d_in[7];
  const float* gk = (const float*)d_in[8];
  const float* bk = (const float*)d_in[9];
  const float* mk = (const float*)d_in[10];
  const float* vk = (const float*)d_in[11];
  const float* wv = (const float*)d_in[12];
  const float* gv = (const float*)d_in[13];
  const float* bv = (const float*)d_in[14];
  const float* mv = (const float*)d_in[15];
  const float* vv = (const float*)d_in[16];
  const float* wg = (const float*)d_in[17];
  const float* gg = (const float*)d_in[18];
  const float* bg = (const float*)d_in[19];
  const float* mg = (const float*)d_in[20];
  const float* vg = (const float*)d_in[21];
  float* out = (float*)d_out;

  // workspace: [0..384K) bf16 Wq|Wv|Wg, [384K..896K) kval f32 [512][256]
  unsigned short* wqb  = (unsigned short*)d_ws;
  unsigned short* wvb  = wqb + 65536;
  unsigned short* wgb  = wqb + 131072;
  float*          kval = (float*)((char*)d_ws + 393216);

  conv_weights_kernel<<<768, 256, 0, stream>>>(wq, wv, wg, wqb);
  kvec_kernel<<<512, 256, 0, stream>>>(sparse, wk, gk, bk, mk, vk, kval);
  attn_main_kernel<<<1024, 512, SMEM_BYTES, stream>>>(
      dense, wqb, wvb, wgb,
      gq, bq, mq, vq, gv, bv, mv, vv, gg, bg, mg, vg, kval, out);
}
